// MultiheadSelfAttention_49237505081583
// MI455X (gfx1250) — matrix operand order repaired and checked by reading, not compiled, not run on hardware
//
#include <hip/hip_runtime.h>
#include <hip/hip_bf16.h>
#include <math.h>

// ---------------------------------------------------------------------------
// Types / helpers for CDNA5 WMMA (gfx1250, wave32)
// ---------------------------------------------------------------------------
typedef __bf16 bf16_t;
typedef __attribute__((ext_vector_type(16))) __bf16        v16bf;
typedef __attribute__((ext_vector_type(4)))  __bf16        v4bf;
typedef __attribute__((ext_vector_type(8)))  float         v8f;
typedef __attribute__((ext_vector_type(4)))  float         f32x4;
typedef __attribute__((ext_vector_type(4)))  unsigned int  u32x4;
typedef __attribute__((ext_vector_type(4)))  int           i32x4;
typedef __attribute__((ext_vector_type(8)))  int           i32x8;

union FragU { v16bf v; u32x4 q[2]; };

// A-matrix 16x32 bf16 fragment: lane L holds row M=L%16,
// K = kb+{0..15} contiguous, kb = 16*(L>>4).
// Caller passes p = &row[k0 + kb]; loads 16B at p and 16B at p+8 elems.
__device__ __forceinline__ v16bf load_frag_a(const bf16_t* p) {
  FragU f;
  f.q[0] = *reinterpret_cast<const u32x4*>(p);
  f.q[1] = *reinterpret_cast<const u32x4*>(p + 8);
  return f.v;
}

// B-matrix 32x16 bf16 fragment: lane L holds column N=L%16,
// K = 16*(L>>4) + {0..15} contiguous over B's column (stored contiguously).
__device__ __forceinline__ v16bf load_frag_b(const bf16_t* p) {
  FragU f;
  f.q[0] = *reinterpret_cast<const u32x4*>(p);
  f.q[1] = *reinterpret_cast<const u32x4*>(p + 8);
  return f.v;
}

__device__ __forceinline__ v8f wmma_bf16(v16bf a, v16bf b, v8f c) {
  return __builtin_amdgcn_wmma_f32_16x16x32_bf16(
      false, a, false, b, (short)0, c, false, false);
}

__device__ __forceinline__ float hw16_max(float v) {  // reduce over 16-lane group
  v = fmaxf(v, __shfl_xor(v, 1, 32));
  v = fmaxf(v, __shfl_xor(v, 2, 32));
  v = fmaxf(v, __shfl_xor(v, 4, 32));
  v = fmaxf(v, __shfl_xor(v, 8, 32));
  return v;
}
__device__ __forceinline__ float hw16_sum(float v) {
  v += __shfl_xor(v, 1, 32);
  v += __shfl_xor(v, 2, 32);
  v += __shfl_xor(v, 4, 32);
  v += __shfl_xor(v, 8, 32);
  return v;
}

#define D_MODEL 1024
#define N_HEADS 16
#define D_K     64
#define SEQ     2048
#define BATCH   4
#define ROWS    (BATCH * SEQ)      // 8192
#define NEG_BIG (-1.0e30f)

// ---------------------------------------------------------------------------
// Tensor Data Mover: async DMA of a 2D weight tile (64 rows x 64 cols bf16,
// row stride 1024 elems) from global into LDS.  Descriptor per ISA §8.3/8.4.
// Guarded: falls back to direct global loads if the builtin is absent.
// ---------------------------------------------------------------------------
#if __has_builtin(__builtin_amdgcn_tensor_load_to_lds) && \
    __has_builtin(__builtin_amdgcn_s_wait_tensorcnt)
#define TDM_OK 1
#else
#define TDM_OK 0
#endif

#if TDM_OK
__device__ __forceinline__ void tdm_load_w_tile(const bf16_t* gsrc,
                                                unsigned lds_off) {
  const unsigned long long ga = (unsigned long long)(const void*)gsrc;
  u32x4 g0;
  g0[0] = 1u;                                   // count=1, user mode
  g0[1] = lds_off;                              // lds_addr (bytes)
  g0[2] = (unsigned)(ga & 0xFFFFFFFFu);         // global_addr[31:0]
  g0[3] = (unsigned)((ga >> 32) & 0x01FFFFFFu)  // global_addr[56:32]
          | (2u << 30);                         // type = 2 ("image")
  i32x8 g1;
  g1[0] = 1 << 16;                              // data_size=1 (2B), mask=0
  g1[1] = (int)((1024u & 0xFFFFu) << 16);       // tensor_dim0 = 1024 (lo16)
  g1[2] = (int)((1024u >> 16) | ((8192u & 0xFFFFu) << 16)); // dim0 hi / dim1 lo
  g1[3] = (int)((8192u >> 16) | (64u << 16));   // dim1 hi / tile_dim0 = 64
  g1[4] = 64;                                   // tile_dim1 = 64, tile_dim2 = 0
  g1[5] = 1024;                                 // tensor_dim0_stride lo32
  g1[6] = 0;                                    // stride hi16 / dim1_stride lo16
  g1[7] = 0;
  i32x4 gz = {0, 0, 0, 0};
#if __clang_major__ >= 23
  i32x8 gz8 = {0, 0, 0, 0, 0, 0, 0, 0};
  __builtin_amdgcn_tensor_load_to_lds(g0, g1, gz, gz, gz8, 0);
#else
  __builtin_amdgcn_tensor_load_to_lds(g0, g1, gz, gz, 0);
#endif
}
#endif

// ---------------------------------------------------------------------------
// fp32 -> bf16 conversion (4 elems per thread: b128 load, b64 store)
// ---------------------------------------------------------------------------
__global__ void f32_to_bf16_kernel(const float* __restrict__ in,
                                   bf16_t* __restrict__ out, int n) {
  int i = (blockIdx.x * blockDim.x + threadIdx.x) * 4;
  if (i + 3 < n) {
    f32x4 v = *reinterpret_cast<const f32x4*>(in + i);
    v4bf o;
    o[0] = (bf16_t)v[0]; o[1] = (bf16_t)v[1];
    o[2] = (bf16_t)v[2]; o[3] = (bf16_t)v[3];
    *reinterpret_cast<v4bf*>(out + i) = o;
  }
}

// ---------------------------------------------------------------------------
// WMMA GEMM  C[M=8192, N=1024] = X[M,1024] @ W^T   (W is [N,K] row-major)
// MODE 0: Q  -> RoPE, *0.125, store bf16 [B,H,S,d]
// MODE 1: K  -> RoPE,          store bf16 [B,H,S,d]
// MODE 2: V  ->                store bf16 [B,H,d,S]   (transposed)
// MODE 3: O-proj ->            store f32  [M,N] to d_out
// Block: 256 threads = 8 waves; wave tile 32(M) x 64(N); K step 64.
// W tile (64x64, shared by all 8 waves) staged through LDS by the TDM with
// double buffering: DMA of tile k+64 overlaps 16 WMMAs on tile k.
// Grid: (ROWS/256, 1024/64)
// ---------------------------------------------------------------------------
template <int MODE>
__global__ __launch_bounds__(256)
void gemm_wmma_kernel(const bf16_t* __restrict__ X,
                      const bf16_t* __restrict__ W,
                      float*        __restrict__ outF,
                      bf16_t*       __restrict__ outB,
                      const float*  __restrict__ cos_sin,
                      const int*    __restrict__ token_positions) {
  const int lane = threadIdx.x & 31;
  const int wave = threadIdx.x >> 5;
  const int r15  = lane & 15;
  const int half = lane >> 4;
  const int m0   = blockIdx.x * 256 + wave * 32;
  const int n0   = blockIdx.y * 64;
  const int K    = D_MODEL;

  v8f acc[2][4] = {};

  const bf16_t* arow0 = X + (size_t)(m0 + r15) * K;
  const bf16_t* arow1 = X + (size_t)(m0 + 16 + r15) * K;
  const int akb = half * 16;
  const int bkb = half * 16;

#if TDM_OK
  __shared__ bf16_t bstage[2][64 * 64];
  const unsigned bst_off0 = (unsigned)(uintptr_t)&bstage[0][0];
  const unsigned bst_off1 = (unsigned)(uintptr_t)&bstage[1][0];
  if (wave == 0) tdm_load_w_tile(W + (size_t)n0 * K, bst_off0);  // preload k=0
#else
  const bf16_t* brow0 = W + (size_t)(n0 + 0 * 16 + r15) * K;
  const bf16_t* brow1 = W + (size_t)(n0 + 1 * 16 + r15) * K;
  const bf16_t* brow2 = W + (size_t)(n0 + 2 * 16 + r15) * K;
  const bf16_t* brow3 = W + (size_t)(n0 + 3 * 16 + r15) * K;
#endif

  for (int k = 0; k < K; k += 64) {
#if TDM_OK
    const int buf = (k >> 6) & 1;
    __syncthreads();  // previous iteration's readers of buf^1 are done
    if (wave == 0) {
      if (k + 64 < K) {
        tdm_load_w_tile(W + (size_t)n0 * K + (k + 64), buf ? bst_off0 : bst_off1);
        __builtin_amdgcn_s_wait_tensorcnt(1);  // current tile landed
      } else {
        __builtin_amdgcn_s_wait_tensorcnt(0);
      }
    }
    __syncthreads();  // current tile visible to all waves
    const bf16_t* bs = bstage[buf];
#endif
#pragma unroll
    for (int kc = 0; kc < 64; kc += 32) {
      v16bf a0 = load_frag_a(arow0 + k + kc + akb);
      v16bf a1 = load_frag_a(arow1 + k + kc + akb);
#if TDM_OK
      v16bf b0 = load_frag_b(bs + (0 * 16 + r15) * 64 + kc + bkb);
      v16bf b1 = load_frag_b(bs + (1 * 16 + r15) * 64 + kc + bkb);
      v16bf b2 = load_frag_b(bs + (2 * 16 + r15) * 64 + kc + bkb);
      v16bf b3 = load_frag_b(bs + (3 * 16 + r15) * 64 + kc + bkb);
#else
      v16bf b0 = load_frag_b(brow0 + k + kc + bkb);
      v16bf b1 = load_frag_b(brow1 + k + kc + bkb);
      v16bf b2 = load_frag_b(brow2 + k + kc + bkb);
      v16bf b3 = load_frag_b(brow3 + k + kc + bkb);
#endif
      acc[0][0] = wmma_bf16(a0, b0, acc[0][0]);
      acc[1][0] = wmma_bf16(a1, b0, acc[1][0]);
      acc[0][1] = wmma_bf16(a0, b1, acc[0][1]);
      acc[1][1] = wmma_bf16(a1, b1, acc[1][1]);
      acc[0][2] = wmma_bf16(a0, b2, acc[0][2]);
      acc[1][2] = wmma_bf16(a1, b2, acc[1][2]);
      acc[0][3] = wmma_bf16(a0, b3, acc[0][3]);
      acc[1][3] = wmma_bf16(a1, b3, acc[1][3]);
    }
  }

  // Epilogue. C-tile layout: VGPR r, lanes 0-15 -> M=r, lanes 16-31 -> M=8+r,
  // N = lane%16.
#pragma unroll
  for (int ms = 0; ms < 2; ++ms) {
#pragma unroll
    for (int nt = 0; nt < 4; ++nt) {
#pragma unroll
      for (int r = 0; r < 8; ++r) {
        const int m = m0 + ms * 16 + r + half * 8;   // token row 0..8191
        const int n = n0 + nt * 16 + r15;            // feature 0..1023
        float val = acc[ms][nt][r];
        if (MODE == 3) {
          outF[(size_t)m * D_MODEL + n] = val;
        } else {
          const int b  = m >> 11;
          const int s  = m & 2047;
          const int h  = n >> 6;
          const int dd = n & 63;
          if (MODE == 0 || MODE == 1) {
            // RoPE: pairs (even dd, odd dd) sit in adjacent lanes.
            const int pos = token_positions[s];
            const int kk  = dd >> 1;
            const float c  = cos_sin[((size_t)pos * 32 + kk) * 2 + 0];
            const float sn = cos_sin[((size_t)pos * 32 + kk) * 2 + 1];
            const float other = __shfl_xor(val, 1, 32);
            float res;
            if ((dd & 1) == 0) res = c * val - sn * other;   // cos*x0 - sin*x1
            else               res = sn * other + c * val;   // sin*x0 + cos*x1
            if (MODE == 0) res *= 0.125f;                    // fold 1/sqrt(64)
            outB[(((size_t)b * N_HEADS + h) * SEQ + s) * D_K + dd] = (bf16_t)res;
          } else {  // MODE == 2: V transposed [B,H,d,S]
            outB[(((size_t)b * N_HEADS + h) * D_K + dd) * SEQ + s] = (bf16_t)val;
          }
        }
      }
    }
  }
}

// ---------------------------------------------------------------------------
// Flash attention: grid (SEQ/64, B*H), block 128 = 4 waves.
// Each wave: 16 query rows, online softmax over causal kv range in 32-chunks.
// Q pre-scaled by 1/sqrt(d).  V stored transposed for contiguous B-fragments.
// K and V fragments are all loaded at the top of the iteration so the V loads
// stay in flight across the softmax VALU phase.
// Output: bf16 [B, S, H*d]  (A operand of the O-projection GEMM).
// ---------------------------------------------------------------------------
__global__ __launch_bounds__(128)
void flash_attn_kernel(const bf16_t* __restrict__ Q,   // [B,H,S,64]
                       const bf16_t* __restrict__ Kd,  // [B,H,S,64]
                       const bf16_t* __restrict__ Vt,  // [B,H,64,S]
                       bf16_t*       __restrict__ O) { // [B,S,1024]
  __shared__ bf16_t pls[4][16 * 32];   // per-wave P staging tile

  const int lane = threadIdx.x & 31;
  const int wave = threadIdx.x >> 5;
  const int r15  = lane & 15;
  const int half = lane >> 4;
  const int bh   = blockIdx.y;
  const int b    = bh >> 4;
  const int h    = bh & 15;
  const int q0   = blockIdx.x * 64 + wave * 16;

  const bf16_t* Qb = Q  + (size_t)bh * SEQ * D_K;
  const bf16_t* Kb = Kd + (size_t)bh * SEQ * D_K;
  const bf16_t* Vb = Vt + (size_t)bh * D_K * SEQ;
  bf16_t* myP = pls[wave];

  // Q fragments (A-layout), loaded once: d 0..31 and d 32..63
  const bf16_t* qrow = Qb + (size_t)(q0 + r15) * D_K + half * 16;
  const v16bf qf0 = load_frag_a(qrow);
  const v16bf qf1 = load_frag_a(qrow + 32);

  v8f oacc[4] = {};
  float m_run[8], l_run[8];
#pragma unroll
  for (int r = 0; r < 8; ++r) { m_run[r] = NEG_BIG; l_run[r] = 0.0f; }

  const int kv_end = q0 + 16;   // exclusive causal bound for this wave
  for (int kv = 0; kv < kv_end; kv += 32) {
    // Prefetch next kv block's K/V cachelines while we crunch this one.
    if (kv + 32 < kv_end) {
      __builtin_prefetch(Kb + (size_t)(kv + 32 + r15) * D_K, 0, 0);
      __builtin_prefetch(Kb + (size_t)(kv + 48 + r15) * D_K, 0, 0);
      __builtin_prefetch(Vb + (size_t)r15 * SEQ + kv + 32, 0, 0);
    }

    // ---- batch-load all K fragments and all V fragments up front ----
    v16bf kf[2][2];
#pragma unroll
    for (int nt = 0; nt < 2; ++nt) {
      const bf16_t* krow = Kb + (size_t)(kv + nt * 16 + r15) * D_K + half * 16;
      kf[nt][0] = load_frag_b(krow);
      kf[nt][1] = load_frag_b(krow + 32);
    }
    v16bf vf[4];
#pragma unroll
    for (int nt = 0; nt < 4; ++nt) {
      vf[nt] = load_frag_b(Vb + (size_t)(nt * 16 + r15) * SEQ + kv + half * 16);
    }

    // ---- scores: two 16x16 tiles (k = d = 64, two WMMAs each) ----
    v8f st[2];
#pragma unroll
    for (int nt = 0; nt < 2; ++nt) {
      v8f s = {};
      s = wmma_bf16(qf0, kf[nt][0], s);
      s = wmma_bf16(qf1, kf[nt][1], s);
      st[nt] = s;
    }

    // ---- causal mask ----
#pragma unroll
    for (int r = 0; r < 8; ++r) {
      const int qg = q0 + r + half * 8;
      if (kv + r15      > qg) st[0][r] = NEG_BIG;
      if (kv + 16 + r15 > qg) st[1][r] = NEG_BIG;
    }

    // ---- online softmax (a row occupies one 16-lane group) ----
#pragma unroll
    for (int r = 0; r < 8; ++r) {
      float v0 = st[0][r], v1 = st[1][r];
      float mx = hw16_max(fmaxf(v0, v1));
      const float m_new = fmaxf(m_run[r], mx);
      const float scale = __expf(m_run[r] - m_new);
      const float e0 = __expf(v0 - m_new);
      const float e1 = __expf(v1 - m_new);
      const float rs = hw16_sum(e0 + e1);
      l_run[r] = l_run[r] * scale + rs;
      m_run[r] = m_new;
      st[0][r] = e0;
      st[1][r] = e1;
#pragma unroll
      for (int nt = 0; nt < 4; ++nt) oacc[nt][r] *= scale;
    }

    // ---- P: C-layout -> LDS (row-major 16x32 bf16) -> A-layout ----
#pragma unroll
    for (int r = 0; r < 8; ++r) {
      const int row = r + half * 8;
      myP[row * 32 + r15]      = (bf16_t)st[0][r];
      myP[row * 32 + 16 + r15] = (bf16_t)st[1][r];
    }
    asm volatile("s_wait_dscnt 0" ::: "memory");  // same-wave LDS RAW fence

    const v16bf pf = load_frag_a(myP + r15 * 32 + half * 16);

    // ---- O += P @ V   (k = 32 kv, V fragments already resident) ----
#pragma unroll
    for (int nt = 0; nt < 4; ++nt) {
      oacc[nt] = wmma_bf16(pf, vf[nt], oacc[nt]);
    }
    asm volatile("" ::: "memory");  // keep next iter's P stores after pf load
  }

  // ---- normalize and store [B,S,H*d] ----
#pragma unroll
  for (int nt = 0; nt < 4; ++nt) {
#pragma unroll
    for (int r = 0; r < 8; ++r) {
      const int qg = q0 + r + half * 8;
      const int dd = nt * 16 + r15;
      const float val = oacc[nt][r] / l_run[r];
      O[((size_t)b * SEQ + qg) * D_MODEL + h * D_K + dd] = (bf16_t)val;
    }
  }
}

// ---------------------------------------------------------------------------
// Launcher
// ---------------------------------------------------------------------------
extern "C" void kernel_launch(void* const* d_in, const int* in_sizes, int n_in,
                              void* d_out, int out_size, void* d_ws, size_t ws_size,
                              hipStream_t stream) {
  const float* x       = (const float*)d_in[0];
  const float* Wq      = (const float*)d_in[1];
  const float* Wk      = (const float*)d_in[2];
  const float* Wv      = (const float*)d_in[3];
  const float* Wo      = (const float*)d_in[4];
  const float* cos_sin = (const float*)d_in[5];
  const int*   pos     = (const int*)d_in[6];
  float*       out     = (float*)d_out;

  // workspace layout (bf16 elements)
  const size_t nX  = (size_t)ROWS * D_MODEL;               // 8388608
  const size_t nW  = (size_t)D_MODEL * D_MODEL;            // 1048576
  const size_t nQ  = (size_t)BATCH * N_HEADS * SEQ * D_K;  // 8388608
  bf16_t* Xbf = (bf16_t*)d_ws;
  bf16_t* Wqb = Xbf + nX;
  bf16_t* Wkb = Wqb + nW;
  bf16_t* Wvb = Wkb + nW;
  bf16_t* Wob = Wvb + nW;
  bf16_t* Qd  = Wob + nW;     // [B,H,S,64]
  bf16_t* Kd  = Qd  + nQ;     // [B,H,S,64]
  bf16_t* Vtd = Kd  + nQ;     // [B,H,64,S]
  bf16_t* Abf = Vtd + nQ;     // attn out [B,S,1024]

  f32_to_bf16_kernel<<<((int)(nX / 4) + 255) / 256, 256, 0, stream>>>(x,  Xbf, (int)nX);
  f32_to_bf16_kernel<<<((int)(nW / 4) + 255) / 256, 256, 0, stream>>>(Wq, Wqb, (int)nW);
  f32_to_bf16_kernel<<<((int)(nW / 4) + 255) / 256, 256, 0, stream>>>(Wk, Wkb, (int)nW);
  f32_to_bf16_kernel<<<((int)(nW / 4) + 255) / 256, 256, 0, stream>>>(Wv, Wvb, (int)nW);
  f32_to_bf16_kernel<<<((int)(nW / 4) + 255) / 256, 256, 0, stream>>>(Wo, Wob, (int)nW);

  dim3 gg(ROWS / 256, D_MODEL / 64);
  dim3 gb(256);
  gemm_wmma_kernel<0><<<gg, gb, 0, stream>>>(Xbf, Wqb, nullptr, Qd,  cos_sin, pos);
  gemm_wmma_kernel<1><<<gg, gb, 0, stream>>>(Xbf, Wkb, nullptr, Kd,  cos_sin, pos);
  gemm_wmma_kernel<2><<<gg, gb, 0, stream>>>(Xbf, Wvb, nullptr, Vtd, cos_sin, pos);

  flash_attn_kernel<<<dim3(SEQ / 64, BATCH * N_HEADS), 128, 0, stream>>>(Qd, Kd, Vtd, Abf);

  gemm_wmma_kernel<3><<<gg, gb, 0, stream>>>(Abf, Wob, out, nullptr, cos_sin, pos);
}
